// LSA_44220983279789
// MI455X (gfx1250) — matrix operand order repaired and checked by reading, not compiled, not run on hardware
//
#include <hip/hip_runtime.h>
#include <hip/hip_bf16.h>
#include <math.h>

typedef __attribute__((ext_vector_type(16))) __bf16 bf16x16;
typedef __attribute__((ext_vector_type(8)))  __bf16 bf16x8;
typedef __attribute__((ext_vector_type(8)))  float  f32x8;

#define BATCH 8
#define SEQ   1024
#define DIM   512
#define HEADS 8
#define DHEAD 64
#define INNER 512          // HEADS*DHEAD
#define ROWS  (BATCH*SEQ)  // 8192

static __device__ __forceinline__ __bf16 to_bf16(float f) { return (__bf16)f; }

static __device__ __forceinline__ f32x8 wmma_bf16(bf16x16 a, bf16x16 b, f32x8 c) {
  return __builtin_amdgcn_wmma_f32_16x16x32_bf16(false, a, false, b, (short)0, c, false, false);
}

// A fragment, 16x32 bf16 (ISA 7.12.2): lanes 0-15 row M=lane, K in {0..7,16..23};
// lanes 16-31 row M=lane-16, K in {8..15,24..31}. Two 16B loads per lane.
static __device__ __forceinline__ bf16x16 load_fragA(const __bf16* p, int ld) {
  int lane = threadIdx.x & 31;
  int m = lane & 15, half = lane >> 4;
  const __bf16* r0 = p + m * ld + half * 8;
  bf16x8 lo = *(const bf16x8*)(r0);
  bf16x8 hi = *(const bf16x8*)(r0 + 16);
  bf16x16 f;
#pragma unroll
  for (int i = 0; i < 8; ++i) { f[i] = lo[i]; f[i + 8] = hi[i]; }
  return f;
}

// B fragment, 32x16 bf16: p points to a K-major buffer where row n holds the 32
// contiguous K values of output column n. lanes 0-15: K in {0..7,16..23}, lanes 16-31: K in {8..15,24..31}.
static __device__ __forceinline__ bf16x16 load_fragB(const __bf16* p, int ld) {
  int lane = threadIdx.x & 31;
  int n = lane & 15, half = lane >> 4;
  const __bf16* r0 = p + n * ld + half * 8;
  bf16x8 lo = *(const bf16x8*)(r0);
  bf16x8 hi = *(const bf16x8*)(r0 + 16);
  bf16x16 f;
#pragma unroll
  for (int i = 0; i < 8; ++i) { f[i] = lo[i]; f[i + 8] = hi[i]; }
  return f;
}

// ---------------- fp32 -> bf16 conversion kernels ----------------
__global__ void k_cvt(const float* __restrict__ src, __bf16* __restrict__ dst, int n) {
  int i = blockIdx.x * blockDim.x + threadIdx.x;
  if (i < n) dst[i] = to_bf16(src[i]);
}

// src is [R,C] row-major fp32; dst is [C,R] row-major bf16 (transpose).
__global__ void k_cvt_t(const float* __restrict__ src, __bf16* __restrict__ dst, int R, int C) {
  int i = blockIdx.x * blockDim.x + threadIdx.x;
  if (i < R * C) {
    int r = i / C, c = i % C;
    dst[(size_t)c * R + r] = to_bf16(src[i]);
  }
}

// ---------------- tiled WMMA GEMM: C[M,N] = A[M,K] * B[K,N], B given as BT[N,K] ----------------
template <bool OUT_BF16>
__global__ __launch_bounds__(256)
void k_gemm(const __bf16* __restrict__ A, const __bf16* __restrict__ BT,
            void* __restrict__ Cout, int M, int N, int K) {
  const int LDT = 40;  // 32 + 8 pad (80B rows, 16B aligned)
  __shared__ alignas(16) __bf16 As[128 * 40];
  __shared__ alignas(16) __bf16 Bs[128 * 40];

  int tid = threadIdx.x;
  int lane = tid & 31;
  int wid = tid >> 5;
  int waveM = wid & 3;   // 4 waves over M (32 rows each)
  int waveN = wid >> 2;  // 2 waves over N (64 cols each)
  int half = lane >> 4, nin = lane & 15;
  int mBase = blockIdx.x * 128;
  int nBase = blockIdx.y * 128;

  f32x8 acc[2][4];
#pragma unroll
  for (int i = 0; i < 2; ++i)
#pragma unroll
    for (int j = 0; j < 4; ++j) acc[i][j] = (f32x8)0.0f;

  int srow = tid >> 1;            // 0..127
  int sc0 = (tid & 1) * 16;       // 0 or 16

  for (int k0 = 0; k0 < K; k0 += 32) {
    // stage A tile (128 x 32)
    {
      const __bf16* src = A + (size_t)(mBase + srow) * K + k0 + sc0;
      bf16x8 u0 = *(const bf16x8*)(src);
      bf16x8 u1 = *(const bf16x8*)(src + 8);
      *(bf16x8*)&As[srow * LDT + sc0] = u0;
      *(bf16x8*)&As[srow * LDT + sc0 + 8] = u1;
    }
    // stage BT tile (128 x 32)
    {
      const __bf16* src = BT + (size_t)(nBase + srow) * K + k0 + sc0;
      bf16x8 u0 = *(const bf16x8*)(src);
      bf16x8 u1 = *(const bf16x8*)(src + 8);
      *(bf16x8*)&Bs[srow * LDT + sc0] = u0;
      *(bf16x8*)&Bs[srow * LDT + sc0 + 8] = u1;
    }
    __syncthreads();

    bf16x16 aF[2];
#pragma unroll
    for (int i = 0; i < 2; ++i)
      aF[i] = load_fragA(&As[(waveM * 32 + i * 16) * LDT], LDT);
    bf16x16 bF[4];
#pragma unroll
    for (int j = 0; j < 4; ++j)
      bF[j] = load_fragB(&Bs[(waveN * 64 + j * 16) * LDT], LDT);
#pragma unroll
    for (int i = 0; i < 2; ++i)
#pragma unroll
      for (int j = 0; j < 4; ++j) acc[i][j] = wmma_bf16(aF[i], bF[j], acc[i][j]);
    __syncthreads();
  }

  // epilogue: C layout -> VGPR r holds M = r + 8*half, N = lane&15
#pragma unroll
  for (int i = 0; i < 2; ++i) {
#pragma unroll
    for (int j = 0; j < 4; ++j) {
      int gn = nBase + waveN * 64 + j * 16 + nin;
#pragma unroll
      for (int r = 0; r < 8; ++r) {
        int gm = mBase + waveM * 32 + i * 16 + r + 8 * half;
        float v = acc[i][j][r];
        if (OUT_BF16) ((__bf16*)Cout)[(size_t)gm * N + gn] = to_bf16(v);
        else          ((float*)Cout)[(size_t)gm * N + gn] = v;
      }
    }
  }
}

// ---------------- flash attention over one (b,h), 128 query rows per block ----------------
__global__ __launch_bounds__(256)
void k_attn(const __bf16* __restrict__ qkv, __bf16* __restrict__ attn_out,
            const float* __restrict__ temp) {
  const int LDK = 72;  // 64 + 8 pad
  __shared__ alignas(16) __bf16 Kl[64 * 72];        // keys x d
  __shared__ alignas(16) __bf16 Vt[64 * 72];        // d x keys (transposed)
  __shared__ alignas(16) __bf16 Pl[8 * 16 * 72];    // per-wave P staging (16 x 64)

  int tid = threadIdx.x, lane = tid & 31, wid = tid >> 5;
  int half = lane >> 4, nin = lane & 15;
  int bh = blockIdx.x;
  int b = bh >> 3, h = bh & 7;
  int q0 = blockIdx.y * 128 + wid * 16;  // this wave's first query row

  float scale = __expf(temp[0]);
  const __bf16* qbase = qkv + (size_t)(b * SEQ) * (3 * INNER) + h * DHEAD;
  const __bf16* kbase = qbase + INNER;
  const __bf16* vbase = qbase + 2 * INNER;

  // Q fragments (16 x 64 split into two 16x32)
  bf16x16 qA[2];
#pragma unroll
  for (int i = 0; i < 2; ++i)
    qA[i] = load_fragA(qbase + (size_t)q0 * (3 * INNER) + i * 32, 3 * INNER);

  float m_run[8], l_run[8];
  f32x8 oacc[4];
#pragma unroll
  for (int r = 0; r < 8; ++r) { m_run[r] = -3.0e38f; l_run[r] = 0.0f; }
#pragma unroll
  for (int dt = 0; dt < 4; ++dt) oacc[dt] = (f32x8)0.0f;

  __bf16* plw = Pl + wid * 16 * LDK;

  int kr = tid >> 2;            // 0..63 (key row for staging)
  int c0 = (tid & 3) * 16;      // 0/16/32/48 (d column chunk)

  for (int kv = 0; kv < 16; ++kv) {
    // stage K tile (rows = keys, cols = d)
    {
      const __bf16* src = kbase + (size_t)(kv * 64 + kr) * (3 * INNER) + c0;
      bf16x8 u0 = *(const bf16x8*)(src);
      bf16x8 u1 = *(const bf16x8*)(src + 8);
      *(bf16x8*)&Kl[kr * LDK + c0] = u0;
      *(bf16x8*)&Kl[kr * LDK + c0 + 8] = u1;
    }
    // stage V tile transposed (rows = d, cols = keys)
    {
      const __bf16* src = vbase + (size_t)(kv * 64 + kr) * (3 * INNER) + c0;
      bf16x8 u0 = *(const bf16x8*)(src);
      bf16x8 u1 = *(const bf16x8*)(src + 8);
      __bf16 tmp[16];
      *(bf16x8*)&tmp[0] = u0;
      *(bf16x8*)&tmp[8] = u1;
#pragma unroll
      for (int e = 0; e < 16; ++e) Vt[(c0 + e) * LDK + kr] = tmp[e];
    }
    __syncthreads();

    // S = scale * Q K^T, diagonal masked
    f32x8 sc[4];
#pragma unroll
    for (int ct = 0; ct < 4; ++ct) {
      f32x8 s = (f32x8)0.0f;
#pragma unroll
      for (int kk = 0; kk < 2; ++kk)
        s = wmma_bf16(qA[kk], load_fragB(&Kl[ct * 16 * LDK] + kk * 32, LDK), s);
#pragma unroll
      for (int r = 0; r < 8; ++r) {
        float v = s[r] * scale;
        int gq = q0 + r + 8 * half;
        int gk = kv * 64 + ct * 16 + nin;
        if (gq == gk) v = -3.4028235e38f;  // diagonal mask
        s[r] = v;
      }
      sc[ct] = s;
    }

    // online softmax: row max / rescale
    float nm[8], al[8];
#pragma unroll
    for (int r = 0; r < 8; ++r) {
      float v = fmaxf(fmaxf(sc[0][r], sc[1][r]), fmaxf(sc[2][r], sc[3][r]));
      v = fmaxf(v, __shfl_xor(v, 1, 16));
      v = fmaxf(v, __shfl_xor(v, 2, 16));
      v = fmaxf(v, __shfl_xor(v, 4, 16));
      v = fmaxf(v, __shfl_xor(v, 8, 16));
      float mn = fmaxf(m_run[r], v);
      al[r] = __expf(m_run[r] - mn);
      m_run[r] = mn;
      nm[r] = mn;
      l_run[r] *= al[r];
    }
#pragma unroll
    for (int dt = 0; dt < 4; ++dt)
#pragma unroll
      for (int r = 0; r < 8; ++r) oacc[dt][r] *= al[r];

    // P = exp(S - m), stage to per-wave LDS for A-fragment reload
#pragma unroll
    for (int ct = 0; ct < 4; ++ct) {
#pragma unroll
      for (int r = 0; r < 8; ++r) {
        float p = __expf(sc[ct][r] - nm[r]);
        sc[ct][r] = p;
        plw[(r + 8 * half) * LDK + ct * 16 + nin] = to_bf16(p);
      }
    }
#pragma unroll
    for (int r = 0; r < 8; ++r) {
      float s = sc[0][r] + sc[1][r] + sc[2][r] + sc[3][r];
      s += __shfl_xor(s, 1, 16);
      s += __shfl_xor(s, 2, 16);
      s += __shfl_xor(s, 4, 16);
      s += __shfl_xor(s, 8, 16);
      l_run[r] += s;
    }

    // wave-local LDS RAW: stores above -> fragment loads below
    asm volatile("s_wait_dscnt 0" ::: "memory");

    bf16x16 pA[2];
#pragma unroll
    for (int kk = 0; kk < 2; ++kk) pA[kk] = load_fragA(plw + kk * 32, LDK);
#pragma unroll
    for (int dt = 0; dt < 4; ++dt) {
#pragma unroll
      for (int kk = 0; kk < 2; ++kk)
        oacc[dt] = wmma_bf16(pA[kk], load_fragB(&Vt[dt * 16 * LDK] + kk * 32, LDK), oacc[dt]);
    }
    __syncthreads();  // before next tile overwrites Kl/Vt
  }

  // normalize and write out in [b, n, h*d] layout (bf16, feeds output GEMM)
#pragma unroll
  for (int dt = 0; dt < 4; ++dt) {
#pragma unroll
    for (int r = 0; r < 8; ++r) {
      int row = q0 + r + 8 * half;
      int col = h * DHEAD + dt * 16 + nin;
      attn_out[(size_t)(b * SEQ + row) * INNER + col] = to_bf16(oacc[dt][r] / l_run[r]);
    }
  }
}

extern "C" void kernel_launch(void* const* d_in, const int* in_sizes, int n_in,
                              void* d_out, int out_size, void* d_ws, size_t ws_size,
                              hipStream_t stream) {
  const float* x      = (const float*)d_in[0];  // [8,1024,512]
  const float* w_qkv  = (const float*)d_in[1];  // [512,1536]
  const float* w_out  = (const float*)d_in[2];  // [512,512]
  const float* temp   = (const float*)d_in[3];  // [1]
  float* out = (float*)d_out;                   // [8,1024,512] fp32

  __bf16* xb    = (__bf16*)d_ws;                        // 8192*512
  __bf16* wqkvT = xb    + (size_t)ROWS * DIM;           // 1536*512 (transposed)
  __bf16* woutT = wqkvT + (size_t)DIM * (3 * INNER);    // 512*512  (transposed)
  __bf16* qkvb  = woutT + (size_t)INNER * DIM;          // 8192*1536
  __bf16* attnb = qkvb  + (size_t)ROWS * (3 * INNER);   // 8192*512

  int nx = ROWS * DIM;
  k_cvt<<<(nx + 255) / 256, 256, 0, stream>>>(x, xb, nx);
  int nq = DIM * 3 * INNER;
  k_cvt_t<<<(nq + 255) / 256, 256, 0, stream>>>(w_qkv, wqkvT, DIM, 3 * INNER);
  int no = INNER * DIM;
  k_cvt_t<<<(no + 255) / 256, 256, 0, stream>>>(w_out, woutT, INNER, DIM);

  // QKV = X * Wqkv   (8192 x 1536, K=512) -> bf16
  k_gemm<true><<<dim3(ROWS / 128, (3 * INNER) / 128), 256, 0, stream>>>(
      xb, wqkvT, (void*)qkvb, ROWS, 3 * INNER, DIM);

  // attention per (b,h), 128 query rows per block
  k_attn<<<dim3(BATCH * HEADS, SEQ / 128), 256, 0, stream>>>(qkvb, attnb, temp);

  // OUT = attn * Wout (8192 x 512, K=512) -> fp32
  k_gemm<false><<<dim3(ROWS / 128, DIM / 128), 256, 0, stream>>>(
      attnb, woutT, (void*)out, ROWS, DIM, INNER);
}
